// MOELayer_14783277433022
// MI455X (gfx1250) — compile-verified, matrix operand order checked by reading, not run on hardware
//
#include <hip/hip_runtime.h>

#define S_TOK 8192
#define MDIM  2048
#define HDIM  4096
#define NEXP  8
#define CAPACITY 2048   // TOP_K * ceil(S/E) = 2*1024
#define NB 8            // routing blocks of 1024 tokens

typedef __attribute__((ext_vector_type(16))) __bf16 v16bf;
typedef __attribute__((ext_vector_type(8)))  float  v8f;
typedef __attribute__((ext_vector_type(4)))  int    v4i;

union FragAB { v16bf v; unsigned int u[8]; };
union FragC  { v8f   v; float        f[8]; };

#if defined(__AMDGCN__)
#define HAS_ASYNC_LDS __has_builtin(__builtin_amdgcn_global_load_async_to_lds_b128)
#else
#define HAS_ASYNC_LDS 0
#endif

__device__ __forceinline__ void wait_async_zero() {
#if defined(__AMDGCN__) && HAS_ASYNC_LDS
  asm volatile("s_wait_asynccnt 0x0" ::: "memory");
#endif
}

__device__ __forceinline__ unsigned short f32_bf16(float f) {
  unsigned int u = __float_as_uint(f);
  u += 0x7fffu + ((u >> 16) & 1u);       // round-to-nearest-even
  return (unsigned short)(u >> 16);
}

__device__ __forceinline__ unsigned int pack_bf16x2(float lo, float hi) {
  return (unsigned int)f32_bf16(lo) | ((unsigned int)f32_bf16(hi) << 16);
}

// ---------------------------------------------------------------- utilities
__global__ void zero16_kernel(uint4* __restrict__ p, size_t n) {
  size_t i  = (size_t)blockIdx.x * blockDim.x + threadIdx.x;
  size_t st = (size_t)gridDim.x * blockDim.x;
  uint4 z; z.x = z.y = z.z = z.w = 0u;
  for (; i < n; i += st) p[i] = z;
}

// ---------------------------------------------------------------- gating
__global__ __launch_bounds__(256) void gate_kernel(
    const float* __restrict__ x, const float* __restrict__ wg,
    int* __restrict__ idx, float* __restrict__ gw,
    float* __restrict__ me_part, int* __restrict__ ce)
{
  const int tid = threadIdx.x;
  const int s   = blockIdx.x * 256 + tid;

  float acc[NEXP];
#pragma unroll
  for (int e = 0; e < NEXP; ++e) acc[e] = 0.f;

  const float4* xr = (const float4*)(x + (size_t)s * MDIM);
  for (int m = 0; m < MDIM / 4; ++m) {
    float4 xv = xr[m];
    const float* w = wg + (size_t)m * 4 * NEXP;   // uniform -> scalar loads
#pragma unroll
    for (int e = 0; e < NEXP; ++e) acc[e] += xv.x * w[e];
#pragma unroll
    for (int e = 0; e < NEXP; ++e) acc[e] += xv.y * w[NEXP + e];
#pragma unroll
    for (int e = 0; e < NEXP; ++e) acc[e] += xv.z * w[2 * NEXP + e];
#pragma unroll
    for (int e = 0; e < NEXP; ++e) acc[e] += xv.w * w[3 * NEXP + e];
  }

  // softmax over 8 experts
  float mx = acc[0];
#pragma unroll
  for (int e = 1; e < NEXP; ++e) mx = fmaxf(mx, acc[e]);
  float gates[NEXP]; float sum = 0.f;
#pragma unroll
  for (int e = 0; e < NEXP; ++e) { gates[e] = __expf(acc[e] - mx); sum += gates[e]; }
  float inv = 1.f / sum;
#pragma unroll
  for (int e = 0; e < NEXP; ++e) gates[e] *= inv;

  // top-2 on logits (first-occurrence ties, like lax.top_k)
  int i0 = 0; float v0 = acc[0];
#pragma unroll
  for (int e = 1; e < NEXP; ++e) if (acc[e] > v0) { v0 = acc[e]; i0 = e; }
  int i1 = -1; float v1 = -3.4e38f;
#pragma unroll
  for (int e = 0; e < NEXP; ++e) if (e != i0 && acc[e] > v1) { v1 = acc[e]; i1 = e; }

  float g0 = gates[i0], g1 = gates[i1];
  float den = fmaxf(g0 + g1, 1.1920929e-07f);
  idx[s * 2] = i0; idx[s * 2 + 1] = i1;
  gw[s * 2]  = g0 / den; gw[s * 2 + 1] = g1 / den;
  atomicAdd(&ce[i0], 1);   // integer atomic: deterministic result

  // deterministic tree reduction of gates -> me partials
  __shared__ float red[256];
#pragma unroll
  for (int e = 0; e < NEXP; ++e) {
    red[tid] = gates[e]; __syncthreads();
    for (int o = 128; o > 0; o >>= 1) { if (tid < o) red[tid] += red[tid + o]; __syncthreads(); }
    if (tid == 0) me_part[blockIdx.x * NEXP + e] = red[0];
    __syncthreads();
  }
}

// ---------------------------------------------------------------- routing
__global__ __launch_bounds__(1024) void count_kernel(
    const int* __restrict__ idx, int* __restrict__ counts)
{
  __shared__ int cnt[16];
  int tid = threadIdx.x;
  if (tid < 16) cnt[tid] = 0;
  __syncthreads();
  int s = blockIdx.x * 1024 + tid;
  atomicAdd(&cnt[0 * 8 + idx[s * 2 + 0]], 1);
  atomicAdd(&cnt[1 * 8 + idx[s * 2 + 1]], 1);
  __syncthreads();
  if (tid < 16) counts[blockIdx.x * 16 + tid] = cnt[tid];
}

__global__ void scan_kernel(const int* __restrict__ counts, int* __restrict__ offs)
{
  __shared__ int tot0[8];
  int t = threadIdx.x;          // blockDim = 16
  if (t < 16) {
    int run = 0;
    for (int b = 0; b < NB; ++b) { offs[b * 16 + t] = run; run += counts[b * 16 + t]; }
    if (t < 8) tot0[t] = run;   // total count of k=0 per expert
  }
  __syncthreads();
  if (t >= 8 && t < 16) {       // k=1 locations start after all k=0 assignments
    int add = tot0[t - 8];
    for (int b = 0; b < NB; ++b) offs[b * 16 + t] += add;
  }
}

__global__ __launch_bounds__(1024) void assign_kernel(
    const int* __restrict__ idx, const int* __restrict__ offs, int* __restrict__ locs)
{
  __shared__ int wcnt[32 * 8];
  __shared__ int wpre[32 * 8];
  const int tid  = threadIdx.x;
  const int lane = tid & 31;
  const int wid  = tid >> 5;
  const int s    = blockIdx.x * 1024 + tid;
#pragma unroll
  for (int k = 0; k < 2; ++k) {
    int e = idx[s * 2 + k];
    unsigned int myb = 0u;
#pragma unroll
    for (int q = 0; q < NEXP; ++q) {
      unsigned int b = (unsigned int)__ballot(e == q);
      if (lane == 0) wcnt[wid * 8 + q] = __popc(b);
      if (e == q) myb = b;
    }
    int rank = __popc(myb & ((1u << lane) - 1u));
    __syncthreads();
    if (tid < 8) {
      int run = 0;
      for (int w = 0; w < 32; ++w) { wpre[w * 8 + tid] = run; run += wcnt[w * 8 + tid]; }
    }
    __syncthreads();
    locs[s * 2 + k] = offs[blockIdx.x * 16 + k * 8 + e] + wpre[wid * 8 + e] + rank;
    __syncthreads();
  }
}

// ---------------------------------------------------------------- dispatch (fp32 -> bf16)
__global__ __launch_bounds__(256) void dispatch_kernel(
    const float* __restrict__ x, const int* __restrict__ idx,
    const int* __restrict__ locs, unsigned short* __restrict__ xbf)
{
  int sk = blockIdx.x; int s = sk >> 1; int k = sk & 1;
  int loc = locs[s * 2 + k];
  if (loc >= CAPACITY) return;                 // dropped token
  int e = idx[s * 2 + k];
  const float2* src = (const float2*)(x + (size_t)s * MDIM);
  unsigned int* dst = (unsigned int*)(xbf + ((size_t)e * CAPACITY + loc) * MDIM);
  for (int i = threadIdx.x; i < MDIM / 2; i += 256) {
    float2 f = src[i];
    dst[i] = pack_bf16x2(f.x, f.y);
  }
}

// ---------------------------------------------------------------- bf16 WMMA GEMM
// C[e] = A[e](Mrows x K, bf16) * B[e](K x N, fp32 converted to bf16 in LDS) + bias
// Ping-pong double-buffered LDS (BK=32 per buffer). A tiles are staged with
// GLOBAL_LOAD_ASYNC_TO_LDS_B128 (CDNA5 async DMA, ASYNCcnt) -> no VGPR staging
// and no loadcnt stalls; B tiles (need fp32->bf16 conversion) issue all global
// loads before any LDS store to keep them in flight together.
// Row strides padded to 56 bf16 (112 B): 16B-aligned (ds_load_b128 merging)
// and 28-dword stride -> 16 distinct LDS banks across lanes.
// RELU_BF16_OUT: fuse ReLU, store bf16 (FC1);  else store fp32 (FC2).
template<bool RELU_BF16_OUT>
__global__ __launch_bounds__(256) void gemm_bf16(
    const unsigned short* __restrict__ A,
    const float* __restrict__ Bsrc,
    const float* __restrict__ bias,
    void* __restrict__ Out,
    int Mrows, int N, int K)
{
  __shared__ __align__(16) unsigned short As0[128 * 56];
  __shared__ __align__(16) unsigned short Bs0[128 * 56];
  __shared__ __align__(16) unsigned short As1[128 * 56];
  __shared__ __align__(16) unsigned short Bs1[128 * 56];

  const int tid  = threadIdx.x;
  const int bx   = blockIdx.x, by = blockIdx.y, e = blockIdx.z;
  const int lane = tid & 31, wid = tid >> 5;
  const int half = lane >> 4, nloc = lane & 15;
  const int wave_m = wid >> 2, wave_n = wid & 3;   // 2x4 waves, each 64x32 out

  const unsigned short* Ae = A    + (size_t)e * Mrows * K;
  const float*          Be = Bsrc + (size_t)e * K * N;
  const float*          be = bias + (size_t)e * N;

  // staging coordinates
  const int arow = tid >> 2;          // A: rows arow, arow+64 (16B per row chunk)
  const int acol = (tid & 3) * 8;     // A: 8 bf16 per chunk
  const int bc   = (tid & 31) * 4;    // B: 4 columns
  const int brh0 = tid >> 5;          // B: k-pair rows brh0, brh0+8

  FragC acc[4][2];
#pragma unroll
  for (int tm = 0; tm < 4; ++tm)
#pragma unroll
    for (int tn = 0; tn < 2; ++tn)
#pragma unroll
      for (int r = 0; r < 8; ++r) acc[tm][tn].f[r] = 0.f;

  auto stage = [&](unsigned short* As_, unsigned short* Bs_, int kt) {
    // ---- A tile: bf16 global -> LDS
#if HAS_ASYNC_LDS
#pragma unroll
    for (int p = 0; p < 2; ++p) {
      const unsigned short* gp = Ae + (size_t)(by * 128 + arow + p * 64) * K + kt + acol;
      unsigned short* lp = As_ + (arow + p * 64) * 56 + acol;
      __builtin_amdgcn_global_load_async_to_lds_b128(
          (__attribute__((address_space(1))) v4i*)gp,
          (__attribute__((address_space(3))) v4i*)lp, 0, 0);
    }
#else
    uint4 areg[2];
#pragma unroll
    for (int p = 0; p < 2; ++p)
      areg[p] = *(const uint4*)(Ae + (size_t)(by * 128 + arow + p * 64) * K + kt + acol);
#pragma unroll
    for (int p = 0; p < 2; ++p)
      *(uint4*)(As_ + (arow + p * 64) * 56 + acol) = areg[p];
#endif
    // ---- B tile: issue all global loads first, then all packed stores
    float4 f[2][2];
#pragma unroll
    for (int p = 0; p < 2; ++p) {
      int krow = kt + 2 * (brh0 + p * 8);
      f[p][0] = *(const float4*)(Be + (size_t)(krow + 0) * N + bx * 128 + bc);
      f[p][1] = *(const float4*)(Be + (size_t)(krow + 1) * N + bx * 128 + bc);
    }
    unsigned int* BsU_ = (unsigned int*)Bs_;
#pragma unroll
    for (int p = 0; p < 2; ++p) {
      int rbh = brh0 + p * 8;             // k-pair index 0..15
      BsU_[(bc + 0) * 28 + rbh] = pack_bf16x2(f[p][0].x, f[p][1].x);
      BsU_[(bc + 1) * 28 + rbh] = pack_bf16x2(f[p][0].y, f[p][1].y);
      BsU_[(bc + 2) * 28 + rbh] = pack_bf16x2(f[p][0].z, f[p][1].z);
      BsU_[(bc + 3) * 28 + rbh] = pack_bf16x2(f[p][0].w, f[p][1].w);
    }
  };

  auto compute = [&](const unsigned short* As_, const unsigned short* Bs_) {
    const unsigned int* AsU_ = (const unsigned int*)As_;
    const unsigned int* BsU_ = (const unsigned int*)Bs_;
    FragAB a[4], b[2];
#pragma unroll
    for (int tn = 0; tn < 2; ++tn) {
      const unsigned int* bp = BsU_ + (wave_n * 32 + tn * 16 + nloc) * 28;
#pragma unroll
      for (int j = 0; j < 8; ++j) {
        int kk = ((j < 4) ? 0 : 16) + half * 8 + (j & 3) * 2;
        b[tn].u[j] = bp[kk >> 1];
      }
    }
#pragma unroll
    for (int tm = 0; tm < 4; ++tm) {
      const unsigned int* ap = AsU_ + (wave_m * 64 + tm * 16 + nloc) * 28;
#pragma unroll
      for (int j = 0; j < 8; ++j) {
        int kk = ((j < 4) ? 0 : 16) + half * 8 + (j & 3) * 2;
        a[tm].u[j] = ap[kk >> 1];
      }
    }
#pragma unroll
    for (int tm = 0; tm < 4; ++tm)
#pragma unroll
      for (int tn = 0; tn < 2; ++tn)
        acc[tm][tn].v = __builtin_amdgcn_wmma_f32_16x16x32_bf16(
            false, a[tm].v, false, b[tn].v, (short)0, acc[tm][tn].v, false, false);
  };

  stage(As0, Bs0, 0);
  wait_async_zero();
  __syncthreads();
  for (int kt = 0; kt < K; kt += 64) {          // K is a multiple of 64
    if (kt + 32 < K) stage(As1, Bs1, kt + 32);  // overlaps with compute below
    compute(As0, Bs0);
    wait_async_zero();                          // drain this wave's async DMAs
    __syncthreads();
    if (kt + 64 < K) stage(As0, Bs0, kt + 64);
    compute(As1, Bs1);
    wait_async_zero();
    __syncthreads();
  }

  // epilogue: C layout per ISA (lanes 0-15: M=r, lanes 16-31: M=8+r; N=lane&15)
  const size_t obase = (size_t)e * Mrows * N;
#pragma unroll
  for (int tn = 0; tn < 2; ++tn) {
    int colg = bx * 128 + wave_n * 32 + tn * 16 + nloc;
    float bv = be[colg];
#pragma unroll
    for (int tm = 0; tm < 4; ++tm) {
      int row0 = by * 128 + wave_m * 64 + tm * 16 + half * 8;
#pragma unroll
      for (int r = 0; r < 8; ++r) {
        float v = acc[tm][tn].f[r] + bv;
        size_t o = obase + (size_t)(row0 + r) * N + colg;
        if (RELU_BF16_OUT) {
          ((unsigned short*)Out)[o] = f32_bf16(fmaxf(v, 0.f));
        } else {
          ((float*)Out)[o] = v;
        }
      }
    }
  }
}

// ---------------------------------------------------------------- combine + loss
__global__ __launch_bounds__(256) void combine_kernel(
    const float* __restrict__ eo, const int* __restrict__ idx,
    const int* __restrict__ locs, const float* __restrict__ gw,
    float* __restrict__ y)
{
  int s  = blockIdx.x;
  int i0 = idx[s * 2], i1 = idx[s * 2 + 1];
  int l0 = locs[s * 2], l1 = locs[s * 2 + 1];
  float g0 = (l0 < CAPACITY) ? gw[s * 2]     : 0.f;
  float g1 = (l1 < CAPACITY) ? gw[s * 2 + 1] : 0.f;
  if (l0 >= CAPACITY) l0 = 0;
  if (l1 >= CAPACITY) l1 = 0;
  const float* r0 = eo + ((size_t)i0 * CAPACITY + l0) * MDIM;
  const float* r1 = eo + ((size_t)i1 * CAPACITY + l1) * MDIM;
  float* yo = y + (size_t)s * MDIM;
  for (int m = threadIdx.x; m < MDIM; m += 256)
    yo[m] = g0 * r0[m] + g1 * r1[m];
}

__global__ void loss_kernel(const float* __restrict__ me_part,
                            const int* __restrict__ ce, float* __restrict__ out_loss)
{
  if (threadIdx.x == 0 && blockIdx.x == 0) {
    float loss = 0.f;
    for (int e = 0; e < NEXP; ++e) {
      float me = 0.f;
      for (int b = 0; b < S_TOK / 256; ++b) me += me_part[b * NEXP + e];
      loss += me * (float)ce[e];
    }
    *out_loss = loss * ((float)NEXP / ((float)S_TOK * (float)S_TOK));
  }
}

// ---------------------------------------------------------------- launcher
extern "C" void kernel_launch(void* const* d_in, const int* in_sizes, int n_in,
                              void* d_out, int out_size, void* d_ws, size_t ws_size,
                              hipStream_t stream)
{
  (void)in_sizes; (void)n_in; (void)out_size; (void)ws_size;
  const float* x     = (const float*)d_in[0];
  const float* wg    = (const float*)d_in[1];
  const float* fc1_w = (const float*)d_in[2];
  const float* fc1_b = (const float*)d_in[3];
  const float* fc2_w = (const float*)d_in[4];
  const float* fc2_b = (const float*)d_in[5];
  float* out = (float*)d_out;

  char* ws = (char*)d_ws;
  size_t off = 0;
  auto take = [&](size_t bytes) -> char* {
    char* p = ws + off; off = (off + bytes + 255) & ~(size_t)255; return p;
  };
  int*   idx     = (int*)  take((size_t)S_TOK * 2 * 4);
  float* gw      = (float*)take((size_t)S_TOK * 2 * 4);
  int*   locs    = (int*)  take((size_t)S_TOK * 2 * 4);
  float* me_part = (float*)take((size_t)(S_TOK / 256) * NEXP * 4);
  int*   ce      = (int*)  take(NEXP * 4);
  int*   counts  = (int*)  take(NB * 16 * 4);
  int*   offs    = (int*)  take(NB * 16 * 4);
  unsigned short* xbf = (unsigned short*)take((size_t)NEXP * CAPACITY * MDIM * 2);
  unsigned short* hbf = (unsigned short*)take((size_t)NEXP * CAPACITY * HDIM * 2);
  float* eo      = (float*)take((size_t)NEXP * CAPACITY * MDIM * 4);

  // zero ce (8 ints) and the dispatched-token bf16 buffer (dropped slots stay 0)
  zero16_kernel<<<1, 64, 0, stream>>>((uint4*)ce, (NEXP * 4) / 16);
  zero16_kernel<<<4096, 256, 0, stream>>>((uint4*)xbf,
      ((size_t)NEXP * CAPACITY * MDIM * 2) / 16);

  gate_kernel  <<<S_TOK / 256, 256, 0, stream>>>(x, wg, idx, gw, me_part, ce);
  count_kernel <<<NB, 1024, 0, stream>>>(idx, counts);
  scan_kernel  <<<1, 16, 0, stream>>>(counts, offs);
  assign_kernel<<<NB, 1024, 0, stream>>>(idx, offs, locs);
  dispatch_kernel<<<S_TOK * 2, 256, 0, stream>>>(x, idx, locs, xbf);

  gemm_bf16<true ><<<dim3(HDIM / 128, CAPACITY / 128, NEXP), 256, 0, stream>>>(
      xbf, fc1_w, fc1_b, (void*)hbf, CAPACITY, HDIM, MDIM);
  gemm_bf16<false><<<dim3(MDIM / 128, CAPACITY / 128, NEXP), 256, 0, stream>>>(
      hbf, fc2_w, fc2_b, (void*)eo, CAPACITY, MDIM, HDIM);

  combine_kernel<<<S_TOK, 256, 0, stream>>>(eo, idx, locs, gw, out);
  loss_kernel   <<<1, 32, 0, stream>>>(me_part, ce, out + (size_t)S_TOK * MDIM);
}
